// BiLSTM_CRF_extra_25933012533513
// MI455X (gfx1250) — compile-verified, matrix operand order checked by reading, not run on hardware
//
#include <hip/hip_runtime.h>
#include <hip/hip_bf16.h>

// ---------------------------------------------------------------------------
// BiLSTM-CRF for MI455X (gfx1250, wave32).
//  - Time-parallel GEMMs (input projection, FC) via v_wmma_f32_16x16x32_f16.
//  - Serial LSTM recurrence: one persistent 1024-thread workgroup per
//    direction, h/gates in LDS, f32 state, f16 weight stream (v_fma_mix).
//  - Serial Viterbi: one 64-thread workgroup, transitions staged in LDS.
// ---------------------------------------------------------------------------

typedef __attribute__((ext_vector_type(16))) _Float16 v16h;
typedef __attribute__((ext_vector_type(8)))  float    v8f;

#define S_LEN 2048
#define D_EMB 256
#define D_EXT 64
#define D_IN  320      // D_EMB + D_EXT
#define H_DIM 256
#define G_DIM 1024     // 4*H
#define HID   512
#define NTAG  64

struct alignas(16) H8 { _Float16 h[8]; };

// Load a 16x32 f16 A-fragment (or mirrored B-fragment) for one wave.
// row-major source, leading dim = ld (halves). Per the CDNA5 VGPR tables:
// lanes 0-15 hold K = {kb+0..7, kb+16..23}; lanes 16-31 hold K+8 of each.
__device__ inline v16h load_frag_f16(const _Float16* base, int ld, int row,
                                     int kb, int lane) {
    int khi = (lane >= 16) ? 8 : 0;
    const _Float16* p = base + (size_t)row * ld + kb + khi;
    union { v16h v; H8 part[2]; } u;
    u.part[0] = *(const H8*)(p);
    u.part[1] = *(const H8*)(p + 16);
    return u.v;
}

// ---------------------------------------------------------------------------
// 1. Embedding gather + f32->f16 concat:  xh[S, 320]
// ---------------------------------------------------------------------------
__global__ void embed_gather(const int* __restrict__ sent,
                             const int* __restrict__ extra,
                             const float* __restrict__ emb,
                             const float* __restrict__ xemb,
                             _Float16* __restrict__ xh) {
    int idx = blockIdx.x * blockDim.x + threadIdx.x;
    if (idx >= S_LEN * D_IN) return;
    int t = idx / D_IN, d = idx - t * D_IN;
    float v = (d < D_EMB) ? emb[(size_t)sent[t] * D_EMB + d]
                          : xemb[(size_t)extra[t] * D_EXT + (d - D_EMB)];
    xh[idx] = (_Float16)v;
}

__global__ void f32_to_f16(const float* __restrict__ in,
                           _Float16* __restrict__ out, int n) {
    int i = blockIdx.x * blockDim.x + threadIdx.x;
    if (i < n) out[i] = (_Float16)in[i];
}

// ---------------------------------------------------------------------------
// 2. WMMA GEMM:  out[M, NTOT] = A[M, K] * W[NTOT, K]^T + bias0 (+ bias1)
//    One 16x16 tile per wave, K stepped by 32 through v_wmma_f32_16x16x32_f16.
// ---------------------------------------------------------------------------
template <int K, int NTOT>
__global__ __launch_bounds__(128) void wmma_gemm_bias(
        const _Float16* __restrict__ A, const _Float16* __restrict__ W,
        const float* __restrict__ bias0, const float* __restrict__ bias1,
        float* __restrict__ out) {
    const int lane = threadIdx.x & 31;
    const int wv   = threadIdx.x >> 5;          // 4 waves / block
    const int tm   = blockIdx.x;                // M tile
    const int tn   = blockIdx.y * 4 + wv;       // N tile
    const int m0 = tm * 16, n0 = tn * 16;

    const int arow = m0 + (lane & 15);
    const int brow = n0 + (lane & 15);

    v8f acc = {};
    for (int kb = 0; kb < K; kb += 32) {
        v16h a = load_frag_f16(A, K, arow, kb, lane);
        v16h b = load_frag_f16(W, K, brow, kb, lane);
        acc = __builtin_amdgcn_wmma_f32_16x16x32_f16(
                  false, a, false, b, (short)0, acc, false, false);
    }

    const int col   = n0 + (lane & 15);
    const int rbase = m0 + ((lane >> 4) << 3);
    float bias = bias0[col] + (bias1 ? bias1[col] : 0.0f);
#pragma unroll
    for (int v = 0; v < 8; ++v)
        out[(size_t)(rbase + v) * NTOT + col] = acc[v] + bias;
}

// ---------------------------------------------------------------------------
// 3. Persistent LSTM recurrence, one workgroup per direction.
//    h (256 f32) and gates (1024 f32) in LDS; c in registers of threads
//    0..255. Weights streamed as f16 (halves per-step L2 traffic vs f32);
//    (float)h16 * f32 accumulate lowers to v_fma_mix_f32.
// ---------------------------------------------------------------------------
__device__ inline float sigmoidf_fast(float x) {
    return 1.0f / (1.0f + __expf(-x));
}

__global__ __launch_bounds__(1024, 1) void lstm_scan(
        const float* __restrict__ gx_f, const float* __restrict__ gx_b,
        const _Float16* __restrict__ whh_f_h, const _Float16* __restrict__ whh_b_h,
        _Float16* __restrict__ lstm_out_h) {
    const int dir = blockIdx.x;                 // 0 = fwd, 1 = bwd
    const float* __restrict__ gx        = dir ? gx_b    : gx_f;
    const _Float16* __restrict__ whh_h  = dir ? whh_b_h : whh_f_h;

    __shared__ float hsh[H_DIM];
    __shared__ float gsh[G_DIM];

    const int j = threadIdx.x;                  // 0..1023 (one gate each)
    if (j < H_DIM) hsh[j] = 0.0f;
    float c = 0.0f;
    __syncthreads();

    const H8* wrow = (const H8*)(whh_h + (size_t)j * H_DIM);

    for (int t = 0; t < S_LEN; ++t) {
        const int pos = dir ? (S_LEN - 1 - t) : t;   // gx row & output row
        float acc = gx[(size_t)pos * G_DIM + j];

        // Prefetch next timestep's gx row (global_prefetch_b8).
        if (t + 1 < S_LEN) {
            const int npos = dir ? (pos - 1) : (pos + 1);
            __builtin_prefetch(gx + (size_t)npos * G_DIM + j, 0, 3);
        }

#pragma unroll 4
        for (int k8 = 0; k8 < H_DIM / 8; ++k8) {
            H8 wv = wrow[k8];
            const float* hp = &hsh[k8 * 8];
#pragma unroll
            for (int u = 0; u < 8; ++u)
                acc += (float)wv.h[u] * hp[u];   // v_fma_mix_f32
        }
        gsh[j] = acc;
        __syncthreads();
        if (j < H_DIM) {
            float ig = sigmoidf_fast(gsh[j]);
            float fg = sigmoidf_fast(gsh[H_DIM + j]);
            float gg = tanhf(gsh[2 * H_DIM + j]);
            float og = sigmoidf_fast(gsh[3 * H_DIM + j]);
            c = fg * c + ig * gg;
            float hn = og * tanhf(c);
            hsh[j] = hn;
            lstm_out_h[(size_t)pos * HID + dir * H_DIM + j] = (_Float16)hn;
        }
        __syncthreads();
    }
}

// ---------------------------------------------------------------------------
// 4. Viterbi: forward max-plus scan + backtrace. 64 threads, trans in LDS.
// ---------------------------------------------------------------------------
__global__ __launch_bounds__(64) void viterbi(
        const float* __restrict__ feats, const float* __restrict__ start,
        const float* __restrict__ endv, const float* __restrict__ trans,
        const int* __restrict__ bptr, const int* __restrict__ eptr,
        int* __restrict__ bp, float* __restrict__ out) {
    __shared__ float score[NTAG];
    __shared__ float tr[NTAG * NTAG];

    const int j = threadIdx.x;
    for (int i = j; i < NTAG * NTAG; i += NTAG) tr[i] = trans[i];

    const int b = bptr[0], e = eptr[0];
    const int L = e - b;
    const float* f0 = feats + (size_t)b * NTAG;

    score[j] = start[j] + f0[j];
    __syncthreads();

    for (int t = 1; t < L; ++t) {
        float emit = f0[(size_t)t * NTAG + j];
        float best = -3.4e38f;
        int   arg  = 0;
#pragma unroll 8
        for (int i = 0; i < NTAG; ++i) {
            float v = score[i] + tr[i * NTAG + j];
            if (v > best) { best = v; arg = i; }   // '>' keeps first max (jnp.argmax)
        }
        bp[(size_t)(t - 1) * NTAG + j] = arg;
        __syncthreads();
        score[j] = best + emit;
        __syncthreads();
    }

    score[j] += endv[j];
    __syncthreads();

    if (j == 0) {
        float best = score[0]; int tag = 0;
        for (int i = 1; i < NTAG; ++i)
            if (score[i] > best) { best = score[i]; tag = i; }
        out[L - 1] = (float)tag;
        for (int t = L - 2; t >= 0; --t) {
            tag = bp[(size_t)t * NTAG + tag];
            out[t] = (float)tag;
        }
    }
}

// ---------------------------------------------------------------------------
// Launch
// ---------------------------------------------------------------------------
extern "C" void kernel_launch(void* const* d_in, const int* in_sizes, int n_in,
                              void* d_out, int out_size, void* d_ws, size_t ws_size,
                              hipStream_t stream) {
    const int* sent      = (const int*)d_in[0];
    const int* extra     = (const int*)d_in[1];
    const int* bptr      = (const int*)d_in[2];
    const int* eptr      = (const int*)d_in[3];
    const float* emb     = (const float*)d_in[4];
    const float* xemb    = (const float*)d_in[5];
    const float* w_ih_f  = (const float*)d_in[6];
    const float* w_hh_f  = (const float*)d_in[7];
    const float* b_ih_f  = (const float*)d_in[8];
    const float* b_hh_f  = (const float*)d_in[9];
    const float* w_ih_b  = (const float*)d_in[10];
    const float* w_hh_b  = (const float*)d_in[11];
    const float* b_ih_b  = (const float*)d_in[12];
    const float* b_hh_b  = (const float*)d_in[13];
    const float* fc_w    = (const float*)d_in[14];
    const float* fc_b    = (const float*)d_in[15];
    const float* crf_s   = (const float*)d_in[16];
    const float* crf_e   = (const float*)d_in[17];
    const float* crf_tr  = (const float*)d_in[18];

    // Workspace carve-up (~24 MB total).
    char* w = (char*)d_ws;
    size_t off = 0;
    auto carve = [&](size_t bytes) -> void* {
        void* p = w + off;
        off += (bytes + 255) & ~(size_t)255;
        return p;
    };
    _Float16* xh      = (_Float16*)carve((size_t)S_LEN * D_IN * 2);
    _Float16* wih_f_h = (_Float16*)carve((size_t)G_DIM * D_IN * 2);
    _Float16* wih_b_h = (_Float16*)carve((size_t)G_DIM * D_IN * 2);
    _Float16* whh_f_h = (_Float16*)carve((size_t)G_DIM * H_DIM * 2);
    _Float16* whh_b_h = (_Float16*)carve((size_t)G_DIM * H_DIM * 2);
    _Float16* fcw_h   = (_Float16*)carve((size_t)NTAG * HID * 2);
    float*    gx_f    = (float*)carve((size_t)S_LEN * G_DIM * 4);
    float*    gx_b    = (float*)carve((size_t)S_LEN * G_DIM * 4);
    _Float16* lstm_h  = (_Float16*)carve((size_t)S_LEN * HID * 2);
    float*    feats   = (float*)carve((size_t)S_LEN * NTAG * 4);
    int*      bp      = (int*)carve((size_t)S_LEN * NTAG * 4);
    (void)ws_size; (void)in_sizes; (void)n_in; (void)out_size;

    // 1. Gather + convert inputs/weights to f16.
    embed_gather<<<(S_LEN * D_IN + 255) / 256, 256, 0, stream>>>(
        sent, extra, emb, xemb, xh);
    f32_to_f16<<<(G_DIM * D_IN + 255) / 256, 256, 0, stream>>>(
        w_ih_f, wih_f_h, G_DIM * D_IN);
    f32_to_f16<<<(G_DIM * D_IN + 255) / 256, 256, 0, stream>>>(
        w_ih_b, wih_b_h, G_DIM * D_IN);
    f32_to_f16<<<(G_DIM * H_DIM + 255) / 256, 256, 0, stream>>>(
        w_hh_f, whh_f_h, G_DIM * H_DIM);
    f32_to_f16<<<(G_DIM * H_DIM + 255) / 256, 256, 0, stream>>>(
        w_hh_b, whh_b_h, G_DIM * H_DIM);
    f32_to_f16<<<(NTAG * HID + 255) / 256, 256, 0, stream>>>(
        fc_w, fcw_h, NTAG * HID);

    // 2. gx = x @ W_ih^T + (b_ih + b_hh), both directions.
    //    M tiles = 2048/16 = 128, N tiles = 1024/16 = 64 (grid.y=16 x 4 waves).
    wmma_gemm_bias<D_IN, G_DIM><<<dim3(S_LEN / 16, G_DIM / 64), 128, 0, stream>>>(
        xh, wih_f_h, b_ih_f, b_hh_f, gx_f);
    wmma_gemm_bias<D_IN, G_DIM><<<dim3(S_LEN / 16, G_DIM / 64), 128, 0, stream>>>(
        xh, wih_b_h, b_ih_b, b_hh_b, gx_b);

    // 3. Serial recurrence, both directions concurrently (grid = 2).
    lstm_scan<<<2, 1024, 0, stream>>>(gx_f, gx_b, whh_f_h, whh_b_h, lstm_h);

    // 4. feats = lstm_out @ fc_w^T + fc_b   (N tiles = 64/16 = 4 -> 4 waves).
    wmma_gemm_bias<HID, NTAG><<<dim3(S_LEN / 16, 1), 128, 0, stream>>>(
        lstm_h, fcw_h, fc_b, (const float*)nullptr, feats);

    // 5. Viterbi decode -> tag path as floats.
    viterbi<<<1, 64, 0, stream>>>(feats, crf_s, crf_e, crf_tr, bptr, eptr,
                                  bp, (float*)d_out);
}
